// ViTBlockClassical_17102559772832
// MI455X (gfx1250) — compile-verified, matrix operand order checked by reading, not run on hardware
//
#include <hip/hip_runtime.h>

// ---------------------------------------------------------------------------
// ViT block (classical) for MI455X / gfx1250.
// wave32 + v_wmma_f32_16x16x32_bf16 + async global->LDS DMA, double-buffered.
// All GEMMs consume B in [N,K] ("transB") layout: weights are transposed at
// the fp32->bf16 conversion step; V is written transposed by its projection.
// ---------------------------------------------------------------------------

typedef __attribute__((ext_vector_type(16))) __bf16   v16bf;
typedef __attribute__((ext_vector_type(8)))  float    v8f;
typedef __attribute__((ext_vector_type(4)))  unsigned v4u;

#define E_DIM   768
#define H_DIM   12
#define DK_DIM  64
#define FFN_DIM 3072
#define B_DIM   32
#define S_DIM   512
#define NTOK    (B_DIM * S_DIM)   // 16384

// GEMM tiling
#define BM 128
#define BN 128
#define BK 32
#define LDS_STRIDE 40   // BK + 8 bf16 pad -> 80B row stride (16B aligned)

union Frag16 {
    v4u   u[2];
    v16bf v;
};

// ---------------------------------------------------------------------------
// CDNA5 async global->LDS helpers (ASYNCcnt-tracked DMA, no VGPR staging)
// ---------------------------------------------------------------------------
__device__ __forceinline__ unsigned lds_byte_off(const void* p)
{
    // generic LDS pointer = {SHARED aperture, lds byte offset} -> low 32 bits
    return (unsigned)(unsigned long long)p;
}

// copy 32 contiguous bytes (16 bf16) per lane: two async b128, INST_OFFSET
// applies to both the LDS and global side.
__device__ __forceinline__ void async_copy_row32(unsigned lds, const __bf16* g)
{
    asm volatile("global_load_async_to_lds_b128 %0, %1, off"
                 :: "v"(lds), "v"(g) : "memory");
    asm volatile("global_load_async_to_lds_b128 %0, %1, off offset:16"
                 :: "v"(lds), "v"(g) : "memory");
}

__device__ __forceinline__ void wait_async0()
{
    asm volatile("s_wait_asynccnt 0x0" ::: "memory");
}

// ---------------------------------------------------------------------------
// fp32 -> bf16 conversion (x)
// ---------------------------------------------------------------------------
__global__ __launch_bounds__(256) void f32_to_bf16_kernel(
    const float* __restrict__ in, __bf16* __restrict__ out, long n)
{
    long i = (long)blockIdx.x * blockDim.x + threadIdx.x;
    if (i < n) out[i] = (__bf16)in[i];
}

// ---------------------------------------------------------------------------
// fp32 [K,N] -> bf16 [N,K] transpose (weights)
// ---------------------------------------------------------------------------
__global__ __launch_bounds__(256) void transpose_f32_to_bf16_kernel(
    const float* __restrict__ src, __bf16* __restrict__ dst, int K, int N)
{
    long i = (long)blockIdx.x * blockDim.x + threadIdx.x;
    if (i < (long)K * N) {
        const int k = (int)(i / N);
        const int n = (int)(i % N);
        dst[(long)n * K + k] = (__bf16)src[i];
    }
}

// ---------------------------------------------------------------------------
// bf16 WMMA GEMM: C[M,N] = A[M,K] * Bt[N,K]^T (+bias)(+relu)
//   Batched via blockIdx.z: off = (z / ic) * so + (z % ic) * si
//   Requirements: K % 32 == 0, M % 128 == 0, N % 16 == 0.
//   OUT_TBF16: store bf16 transposed per 512-row batch: [b][col][row%512]
// ---------------------------------------------------------------------------
template <bool HAS_BIAS, bool RELU, bool OUT_F32, bool OUT_BF16, bool OUT_TBF16>
__global__ __launch_bounds__(256) void gemm_bf16_wmma(
    const __bf16* __restrict__ A, const __bf16* __restrict__ Bt,
    float* __restrict__ Cf, __bf16* __restrict__ Cb,
    const float* __restrict__ bias,
    int M, int N, int K, int lda, int ldb, int ldc,
    int ic, long soA, long siA, long soB, long siB, long soC, long siC)
{
    __shared__ __bf16 As[2][BM * LDS_STRIDE];
    __shared__ __bf16 Bs[2][BN * LDS_STRIDE];

    const int tid  = threadIdx.x;
    const int lane = tid & 31;
    const int wave = tid >> 5;
    const int l16  = lane & 15;
    const int hi   = lane >> 4;          // 0: K-chunks 0-7/16-23, 1: 8-15/24-31
    const int kb   = hi << 3;

    const int wm = (wave & 3) * 32;      // wave row base within block tile
    const int wn = (wave >> 2) * 64;     // wave col base within block tile

    const int bm0 = blockIdx.y * BM;
    const int bn0 = blockIdx.x * BN;

    const int z = blockIdx.z;
    const long offA = (long)(z / ic) * soA + (long)(z % ic) * siA;
    const long offB = (long)(z / ic) * soB + (long)(z % ic) * siB;
    const long offC = (long)(z / ic) * soC + (long)(z % ic) * siC;

    // staging assignment: thread -> (row = tid>>1, 16-elem col chunk = tid&1)
    const int row_ld = tid >> 1;
    const int col_ld = (tid & 1) << 4;
    const __bf16* __restrict__ aptr =
        A + offA + (long)(bm0 + row_ld) * lda + col_ld;
    const __bf16* __restrict__ bptr =
        Bt + offB + (long)(bn0 + row_ld) * ldb + col_ld;
    const bool bok = (bn0 + row_ld) < N;   // EXEC-masks the B async issue

    unsigned lA[2], lB[2];
    lA[0] = lds_byte_off(&As[0][row_ld * LDS_STRIDE + col_ld]);
    lA[1] = lds_byte_off(&As[1][row_ld * LDS_STRIDE + col_ld]);
    lB[0] = lds_byte_off(&Bs[0][row_ld * LDS_STRIDE + col_ld]);
    lB[1] = lds_byte_off(&Bs[1][row_ld * LDS_STRIDE + col_ld]);

    v8f acc[2][4];
#pragma unroll
    for (int r = 0; r < 2; ++r)
#pragma unroll
        for (int c = 0; c < 4; ++c)
            acc[r][c] = (v8f)0.0f;

    // prefetch first K-tile into buffer 0
    async_copy_row32(lA[0], aptr);
    if (bok) async_copy_row32(lB[0], bptr);

    int cur = 0;
    for (int k0 = 0; k0 < K; k0 += BK) {
        wait_async0();      // my async copies into buf[cur] landed
        __syncthreads();    // everyone's copies landed; prev readers done

        // overlap: DMA next K-tile into the other buffer while we compute
        if (k0 + BK < K) {
            async_copy_row32(lA[cur ^ 1], aptr + (k0 + BK));
            if (bok) async_copy_row32(lB[cur ^ 1], bptr + (k0 + BK));
        }

        // fragment loads (ISA 7.12.2 16-bit A/B layout): 2x ds_load_b128 each
        Frag16 fa[2], fb[4];
#pragma unroll
        for (int r = 0; r < 2; ++r) {
            const __bf16* base = &As[cur][(wm + r * 16 + l16) * LDS_STRIDE + kb];
            fa[r].u[0] = *(const v4u*)(base);
            fa[r].u[1] = *(const v4u*)(base + 16);
        }
#pragma unroll
        for (int c = 0; c < 4; ++c) {
            const __bf16* base = &Bs[cur][(wn + c * 16 + l16) * LDS_STRIDE + kb];
            fb[c].u[0] = *(const v4u*)(base);
            fb[c].u[1] = *(const v4u*)(base + 16);
        }

#pragma unroll
        for (int r = 0; r < 2; ++r)
#pragma unroll
            for (int c = 0; c < 4; ++c)
                acc[r][c] = __builtin_amdgcn_wmma_f32_16x16x32_bf16(
                    false, fa[r].v, false, fb[c].v,
                    (short)0, acc[r][c], false, false);

        __syncthreads();    // all waves done reading buf[cur]
        cur ^= 1;
    }

    // ---- epilogue: D layout = VGPR i -> M = base + hi*8 + i, N = base + l16
#pragma unroll
    for (int r = 0; r < 2; ++r) {
        const int row0 = bm0 + wm + r * 16 + hi * 8;
#pragma unroll
        for (int c = 0; c < 4; ++c) {
            const int col = bn0 + wn + c * 16 + l16;
            if (col < N) {
                const float bv = HAS_BIAS ? bias[col] : 0.0f;
#pragma unroll
                for (int i = 0; i < 8; ++i) {
                    float v = acc[r][c][i] + bv;
                    if (RELU) v = fmaxf(v, 0.0f);
                    const int row = row0 + i;
                    if (OUT_F32 || OUT_BF16) {
                        const long idx = offC + (long)row * ldc + col;
                        if (OUT_F32)  Cf[idx] = v;
                        if (OUT_BF16) Cb[idx] = (__bf16)v;
                    }
                    if (OUT_TBF16) {
                        // [b][col][s], b = row/512, s = row%512
                        const long tidx = offC +
                            (((long)(row >> 9) * N) << 9) +
                            ((long)col << 9) + (row & 511);
                        Cb[tidx] = (__bf16)v;
                    }
                }
            }
        }
    }
}

// ---------------------------------------------------------------------------
// Row softmax over 512-wide score rows (scale folded in), bf16 probs out.
// ---------------------------------------------------------------------------
__global__ __launch_bounds__(128) void softmax_rows_kernel(
    const float* __restrict__ scores, __bf16* __restrict__ probs, float scale)
{
    __shared__ float red[128];
    const long row = blockIdx.x;
    const float* s = scores + row * S_DIM;
    __bf16* p = probs + row * S_DIM;
    const int tid = threadIdx.x;

    float v[4];
    float mx = -3.0e38f;
#pragma unroll
    for (int j = 0; j < 4; ++j) {
        v[j] = s[tid + j * 128] * scale;
        mx = fmaxf(mx, v[j]);
    }
    red[tid] = mx;
    __syncthreads();
    for (int off = 64; off > 0; off >>= 1) {
        if (tid < off) red[tid] = fmaxf(red[tid], red[tid + off]);
        __syncthreads();
    }
    mx = red[0];
    __syncthreads();

    float sum = 0.0f;
#pragma unroll
    for (int j = 0; j < 4; ++j) {
        v[j] = __expf(v[j] - mx);
        sum += v[j];
    }
    red[tid] = sum;
    __syncthreads();
    for (int off = 64; off > 0; off >>= 1) {
        if (tid < off) red[tid] += red[tid + off];
        __syncthreads();
    }
    const float inv = 1.0f / red[0];
#pragma unroll
    for (int j = 0; j < 4; ++j)
        p[tid + j * 128] = (__bf16)(v[j] * inv);
}

// ---------------------------------------------------------------------------
// Residual add + LayerNorm over E=768, emits f32 + optional bf16.
// ---------------------------------------------------------------------------
__global__ __launch_bounds__(256) void add_layernorm_kernel(
    const float* __restrict__ a, const float* __restrict__ b,
    const float* __restrict__ g, const float* __restrict__ be,
    float* __restrict__ outf, __bf16* __restrict__ outb)
{
    __shared__ float r1[256];
    __shared__ float r2[256];
    const long row = blockIdx.x;
    const float* pa = a + row * E_DIM;
    const float* pb = b + row * E_DIM;
    const int tid = threadIdx.x;

    float v[3];
    float s = 0.0f, s2 = 0.0f;
#pragma unroll
    for (int j = 0; j < 3; ++j) {
        const int c = tid + j * 256;
        v[j] = pa[c] + pb[c];
        s += v[j];
        s2 += v[j] * v[j];
    }
    r1[tid] = s;
    r2[tid] = s2;
    __syncthreads();
    for (int off = 128; off > 0; off >>= 1) {
        if (tid < off) {
            r1[tid] += r1[tid + off];
            r2[tid] += r2[tid + off];
        }
        __syncthreads();
    }
    const float mu  = r1[0] * (1.0f / E_DIM);
    const float var = r2[0] * (1.0f / E_DIM) - mu * mu;
    const float rs  = rsqrtf(var + 1e-5f);
#pragma unroll
    for (int j = 0; j < 3; ++j) {
        const int c = tid + j * 256;
        const float y = (v[j] - mu) * rs * g[c] + be[c];
        if (outf) outf[row * E_DIM + c] = y;
        if (outb) outb[row * E_DIM + c] = (__bf16)y;
    }
}

// ---------------------------------------------------------------------------
// Host orchestration
// ---------------------------------------------------------------------------
extern "C" void kernel_launch(void* const* d_in, const int* in_sizes, int n_in,
                              void* d_out, int out_size, void* d_ws, size_t ws_size,
                              hipStream_t stream)
{
    (void)in_sizes; (void)n_in; (void)out_size; (void)ws_size;

    const float* x   = (const float*)d_in[0];
    const float* Wq  = (const float*)d_in[1];
    const float* Wk  = (const float*)d_in[2];
    const float* Wv  = (const float*)d_in[3];
    const float* Wo  = (const float*)d_in[4];
    const float* W1  = (const float*)d_in[5];
    const float* b1  = (const float*)d_in[6];
    const float* W2  = (const float*)d_in[7];
    const float* b2  = (const float*)d_in[8];
    const float* g1  = (const float*)d_in[9];
    const float* be1 = (const float*)d_in[10];
    const float* g2  = (const float*)d_in[11];
    const float* be2 = (const float*)d_in[12];

    // ---- workspace carving (256B aligned)
    char* wp = (char*)d_ws;
    auto carve = [&](size_t bytes) -> char* {
        char* r = wp;
        wp += (bytes + 255) & ~(size_t)255;
        return r;
    };
    const long nXE = (long)NTOK * E_DIM;
    const long nEE = (long)E_DIM * E_DIM;
    const long nEF = (long)E_DIM * FFN_DIM;
    const long nBH = (long)B_DIM * H_DIM;
    const long nSS = (long)S_DIM * S_DIM;

    __bf16* xb     = (__bf16*)carve(nXE * 2);
    __bf16* Wqt    = (__bf16*)carve(nEE * 2);  // [N=E, K=E]
    __bf16* Wkt    = (__bf16*)carve(nEE * 2);
    __bf16* Wvt    = (__bf16*)carve(nEE * 2);
    __bf16* Wot    = (__bf16*)carve(nEE * 2);
    __bf16* W1t    = (__bf16*)carve(nEF * 2);  // [N=FFN, K=E]
    __bf16* W2t    = (__bf16*)carve(nEF * 2);  // [N=E, K=FFN]
    __bf16* Qb     = (__bf16*)carve(nXE * 2);  // [B,S,E]
    __bf16* Kb     = (__bf16*)carve(nXE * 2);  // [B,S,E]
    __bf16* Vt     = (__bf16*)carve(nXE * 2);  // [B,E,S] (per-head [DK,S])
    float*  scores = (float*) carve(nBH * nSS * 4);
    __bf16* attn   = (__bf16*)carve(nBH * nSS * 2);
    __bf16* ctxb   = (__bf16*)carve(nXE * 2);  // [B,S,E]
    float*  aout   = (float*) carve(nXE * 4);
    float*  y1     = (float*) carve(nXE * 4);
    __bf16* y1b    = (__bf16*)carve(nXE * 2);
    __bf16* h1b    = (__bf16*)carve((long)NTOK * FFN_DIM * 2);
    float*  ffn2   = (float*) carve(nXE * 4);

    const dim3 blk(256);

    // 0) conversions: x -> bf16 ; weights -> transposed bf16 [N,K]
    f32_to_bf16_kernel<<<dim3((unsigned)((nXE + 255) / 256)), blk, 0, stream>>>(
        x, xb, nXE);
    auto tr = [&](const float* src, __bf16* dst, int K, int N) {
        const long n = (long)K * N;
        transpose_f32_to_bf16_kernel<<<dim3((unsigned)((n + 255) / 256)), blk, 0, stream>>>(
            src, dst, K, N);
    };
    tr(Wq, Wqt, E_DIM, E_DIM);
    tr(Wk, Wkt, E_DIM, E_DIM);
    tr(Wv, Wvt, E_DIM, E_DIM);
    tr(Wo, Wot, E_DIM, E_DIM);
    tr(W1, W1t, E_DIM, FFN_DIM);
    tr(W2, W2t, FFN_DIM, E_DIM);

    // 1) projections: [16384,768] x Wt[768,768]^T
    {
        dim3 grid(E_DIM / BN, NTOK / BM, 1);
        gemm_bf16_wmma<false, false, false, true, false><<<grid, blk, 0, stream>>>(
            xb, Wqt, nullptr, Qb, nullptr, NTOK, E_DIM, E_DIM, E_DIM, E_DIM, E_DIM,
            1, 0, 0, 0, 0, 0, 0);
        gemm_bf16_wmma<false, false, false, true, false><<<grid, blk, 0, stream>>>(
            xb, Wkt, nullptr, Kb, nullptr, NTOK, E_DIM, E_DIM, E_DIM, E_DIM, E_DIM,
            1, 0, 0, 0, 0, 0, 0);
        // V stored transposed: Vt[b][e][s]
        gemm_bf16_wmma<false, false, false, false, true><<<grid, blk, 0, stream>>>(
            xb, Wvt, nullptr, Vt, nullptr, NTOK, E_DIM, E_DIM, E_DIM, E_DIM, S_DIM,
            1, 0, 0, 0, 0, 0, 0);
    }

    // 2) scores = Q * K^T per (b,h): M=N=512, K=64, batch=384
    {
        dim3 grid(S_DIM / BN, S_DIM / BM, (unsigned)nBH);
        gemm_bf16_wmma<false, false, true, false, false><<<grid, blk, 0, stream>>>(
            Qb, Kb, scores, nullptr, nullptr, S_DIM, S_DIM, DK_DIM, E_DIM, E_DIM, S_DIM,
            H_DIM, (long)S_DIM * E_DIM, DK_DIM,
                   (long)S_DIM * E_DIM, DK_DIM,
                   (long)H_DIM * nSS,   nSS);
    }

    // 3) softmax(scores / 8) -> bf16 probs
    softmax_rows_kernel<<<dim3((unsigned)(nBH * S_DIM)), dim3(128), 0, stream>>>(
        scores, attn, 0.125f);

    // 4) ctx = probs * V per (b,h): M=512, N=64, K=512; B = Vt[h] = [DK,S]
    {
        dim3 grid(1, S_DIM / BM, (unsigned)nBH);
        gemm_bf16_wmma<false, false, false, true, false><<<grid, blk, 0, stream>>>(
            attn, Vt, nullptr, ctxb, nullptr, S_DIM, DK_DIM, S_DIM, S_DIM, S_DIM, E_DIM,
            H_DIM, (long)H_DIM * nSS,    nSS,
                   (long)E_DIM * S_DIM,  (long)DK_DIM * S_DIM,
                   (long)S_DIM * E_DIM,  DK_DIM);
    }

    // 5) attn_out = ctx * Wo -> f32
    {
        dim3 grid(E_DIM / BN, NTOK / BM, 1);
        gemm_bf16_wmma<false, false, true, false, false><<<grid, blk, 0, stream>>>(
            ctxb, Wot, aout, nullptr, nullptr, NTOK, E_DIM, E_DIM, E_DIM, E_DIM, E_DIM,
            1, 0, 0, 0, 0, 0, 0);
    }

    // 6) y1 = LN(x + attn_out) -> f32 + bf16
    add_layernorm_kernel<<<dim3(NTOK), blk, 0, stream>>>(x, aout, g1, be1, y1, y1b);

    // 7) h1 = relu(y1 * W1 + b1) -> bf16
    {
        dim3 grid(FFN_DIM / BN, NTOK / BM, 1);
        gemm_bf16_wmma<true, true, false, true, false><<<grid, blk, 0, stream>>>(
            y1b, W1t, nullptr, h1b, b1, NTOK, FFN_DIM, E_DIM, E_DIM, E_DIM, FFN_DIM,
            1, 0, 0, 0, 0, 0, 0);
    }

    // 8) ffn2 = h1 * W2 + b2 -> f32
    {
        dim3 grid(E_DIM / BN, NTOK / BM, 1);
        gemm_bf16_wmma<true, false, true, false, false><<<grid, blk, 0, stream>>>(
            h1b, W2t, ffn2, nullptr, b2, NTOK, E_DIM, FFN_DIM, FFN_DIM, FFN_DIM, E_DIM,
            1, 0, 0, 0, 0, 0, 0);
    }

    // 9) out = LN(y1 + ffn2) -> f32 (d_out)
    add_layernorm_kernel<<<dim3(NTOK), blk, 0, stream>>>(
        y1, ffn2, g2, be2, (float*)d_out, nullptr);
}
